// PtrScorer_24721831756012
// MI455X (gfx1250) — matrix operand order repaired and checked by reading, not compiled, not run on hardware
//
#include <hip/hip_runtime.h>
#include <hip/hip_bf16.h>
#include <float.h>

#define NEG_INF_VAL (-100000000.0f)

typedef __attribute__((ext_vector_type(16))) __bf16 bf16x16;
typedef __attribute__((ext_vector_type(8)))  float  floatx8;

#define ASYNC_LDS_OK __has_builtin(__builtin_amdgcn_global_load_async_to_lds_b128)

// The async-copy builtin takes pointers to 16-byte int vectors in explicit
// address spaces: (v4i addrspace(1)*, v4i addrspace(3)*, imm offset, imm cpol).
typedef int intx4 __attribute__((vector_size(16)));
typedef __attribute__((address_space(1))) intx4 gas_intx4;
typedef __attribute__((address_space(3))) intx4 las_intx4;

__device__ __forceinline__ int f32_ord(float x) {
    int i = __float_as_int(x);
    return (i >= 0) ? i : (i ^ 0x7fffffff);
}
__device__ __forceinline__ float ord_f32(int i) {
    return __int_as_float((i >= 0) ? i : (i ^ 0x7fffffff));
}
__device__ __forceinline__ float sigmoidf(float x) { return 1.0f / (1.0f + expf(-x)); }

__device__ __forceinline__ void wait_async0() {
#if ASYNC_LDS_OK
  #if __has_builtin(__builtin_amdgcn_s_wait_asynccnt)
    __builtin_amdgcn_s_wait_asynccnt(0);
  #else
    asm volatile("s_wait_asynccnt 0x0" ::: "memory");
  #endif
#endif
}

// ---------------------------------------------------------------------------
// Fused dual-output feat GEMM:
//   O0[N,H] = A[N,D] @ W0[D,H],  O1[N,H] = A[N,D] @ W1[D,H]
// BM=128, BN=128, BK=32. Weight tiles (shared by all 8 waves) are staged into
// LDS via gfx1250 async global->LDS b128 copies (ASYNCcnt path). Each wave
// owns 16 rows, holds one bf16 A fragment per K-step and issues 16 WMMAs
// against it (8 column tiles x 2 outputs), so A bytes are amortized 16x and
// A is read only H/BN = 4x total for BOTH GEMMs (~512MB @ 23.3TB/s).
// ---------------------------------------------------------------------------
#define PAN_STRIDE 132   // 128 floats + 4 pad: float4-aligned, spreads banks

__global__ void __launch_bounds__(256)
feat_gemm2_kernel(const float* __restrict__ A,
                  const float* __restrict__ W0, const float* __restrict__ W1,
                  float* __restrict__ O0, float* __restrict__ O1,
                  int D, int H) {
    __shared__ float pan[2][32 * PAN_STRIDE + 4];

    const int tid    = threadIdx.x;
    const int wave   = tid >> 5;
    const int lane   = tid & 31;
    const int lane16 = lane & 15;
    const int hi     = lane >> 4;                    // 0: lanes 0-15, 1: 16-31
    const int n0     = blockIdx.x * 128;
    const int m0     = (blockIdx.y * 8 + wave) * 16;

    const float* arow = A + (size_t)(m0 + lane16) * D;   // A frag: M = lane16

    floatx8 acc0[8], acc1[8];
    #pragma unroll
    for (int i = 0; i < 8; ++i) { acc0[i] = (floatx8){}; acc1[i] = (floatx8){}; }

    for (int k0 = 0; k0 < D; k0 += 32) {
        __syncthreads();   // previous iteration's LDS reads complete
        // Stage both 32x128 f32 weight tiles: 2048 b128 chunks / 256 threads.
        #pragma unroll
        for (int j = 0; j < 8; ++j) {
            int ch  = tid + 256 * j;
            int p   = ch >> 10;
            int rem = ch & 1023;
            int kk  = rem >> 5;        // 0..31  (K row within tile)
            int cc  = rem & 31;        // 0..31  (float4 chunk within row)
            const float* gsrc = (p ? W1 : W0) + (size_t)(k0 + kk) * H + n0 + 4 * cc;
            float*       ldst = &pan[p][kk * PAN_STRIDE + 4 * cc];
#if ASYNC_LDS_OK
            __builtin_amdgcn_global_load_async_to_lds_b128(
                (gas_intx4*)gsrc, (las_intx4*)ldst, 0, 0);
#else
            *(float4*)ldst = *(const float4*)gsrc;
#endif
        }
        wait_async0();
        __syncthreads();

        // A fragment (16x32 bf16): VGPR v packs K = hi*8 + (v&4?16:0) + 2*(v&3)
        __builtin_prefetch(arow + k0 + 64, 0, 1);    // global_prefetch_b8
        bf16x16 a;
        #pragma unroll
        for (int v = 0; v < 8; ++v) {
            int koff = (hi ? 8 : 0) + ((v & 4) ? 16 : 0) + 2 * (v & 3);
            float2 av = *(const float2*)(arow + k0 + koff);
            a[2 * v]     = (__bf16)av.x;
            a[2 * v + 1] = (__bf16)av.y;
        }

        // 16 WMMAs against this A fragment. B frag: column = n0 + nt*16 + lane16,
        // element s = K row in the same order as A (hi*8 + (s&8?16:0) + (s&7)).
        #pragma unroll
        for (int nt = 0; nt < 8; ++nt) {
            const float* bp0 = &pan[0][nt * 16 + lane16];
            const float* bp1 = &pan[1][nt * 16 + lane16];
            bf16x16 b0, b1;
            #pragma unroll
            for (int s = 0; s < 16; ++s) {
                int kb = (hi ? 8 : 0) + ((s & 8) ? 16 : 0) + (s & 7);
                b0[s] = (__bf16)bp0[kb * PAN_STRIDE];
                b1[s] = (__bf16)bp1[kb * PAN_STRIDE];
            }
            acc0[nt] = __builtin_amdgcn_wmma_f32_16x16x32_bf16(false, a, false, b0,
                                                               (short)0, acc0[nt], false, false);
            acc1[nt] = __builtin_amdgcn_wmma_f32_16x16x32_bf16(false, a, false, b1,
                                                               (short)0, acc1[nt], false, false);
        }
    }

    // C/D 16x16 f32: VGPR v -> row m0 + v + hi*8, col tile n0+nt*16+lane16
    #pragma unroll
    for (int nt = 0; nt < 8; ++nt) {
        size_t base = (size_t)(m0 + (hi ? 8 : 0)) * H + n0 + nt * 16 + lane16;
        #pragma unroll
        for (int v = 0; v < 8; ++v) {
            O0[base + (size_t)v * H] = acc0[nt][v];
            O1[base + (size_t)v * H] = acc1[nt][v];
        }
    }
}

// ---------------------------------------------------------------------------
// Copy initial state into workspace.
// ---------------------------------------------------------------------------
__global__ void init_state_kernel(const float* __restrict__ init_i,
                                  const float* __restrict__ init_h,
                                  const float* __restrict__ init_c,
                                  float* __restrict__ xbuf,
                                  float* __restrict__ h0,
                                  float* __restrict__ c0) {
    int tid = threadIdx.x;
    xbuf[tid] = init_i[tid];
    h0[tid]   = init_h[tid];
    c0[tid]   = init_c[tid];
}

// ---------------------------------------------------------------------------
// LSTM cell, single block of H threads; each thread owns one hidden unit.
// ---------------------------------------------------------------------------
__global__ void __launch_bounds__(512)
lstm_kernel(const float* __restrict__ x, const float* __restrict__ h,
            const float* __restrict__ cprev,
            const float* __restrict__ w_ih, const float* __restrict__ w_hh,
            const float* __restrict__ b_ih, const float* __restrict__ b_hh,
            float* __restrict__ hout, float* __restrict__ cout,
            int D, int H) {
    __shared__ float xs[512];
    __shared__ float hs[512];
    int tid = threadIdx.x;
    xs[tid] = x[tid];
    hs[tid] = h[tid];
    __syncthreads();

    float gi = b_ih[tid]         + b_hh[tid];
    float gf = b_ih[H + tid]     + b_hh[H + tid];
    float gg = b_ih[2 * H + tid] + b_hh[2 * H + tid];
    float go = b_ih[3 * H + tid] + b_hh[3 * H + tid];

    for (int k = 0; k < H; ++k) {
        float xv = xs[k], hv = hs[k];
        gi += xv * w_ih[(size_t)tid * D + k]           + hv * w_hh[(size_t)tid * H + k];
        gf += xv * w_ih[(size_t)(H + tid) * D + k]     + hv * w_hh[(size_t)(H + tid) * H + k];
        gg += xv * w_ih[(size_t)(2 * H + tid) * D + k] + hv * w_hh[(size_t)(2 * H + tid) * H + k];
        go += xv * w_ih[(size_t)(3 * H + tid) * D + k] + hv * w_hh[(size_t)(3 * H + tid) * H + k];
    }
    float cn = sigmoidf(gf) * cprev[tid] + sigmoidf(gi) * tanhf(gg);
    cout[tid] = cn;
    hout[tid] = sigmoidf(go) * tanhf(cn);
}

// ---------------------------------------------------------------------------
// qw[j] = sum_k q[k] * W[k,j]
// ---------------------------------------------------------------------------
__global__ void gemv_kernel(const float* __restrict__ q, const float* __restrict__ W,
                            float* __restrict__ qw, int H) {
    int j = blockIdx.x * blockDim.x + threadIdx.x;
    float acc = 0.f;
    for (int k = 0; k < H; ++k) acc += q[k] * W[(size_t)k * H + j];
    qw[j] = acc;
}

// ---------------------------------------------------------------------------
// Reset softmax accumulators.
// ---------------------------------------------------------------------------
__global__ void attn_reset_kernel(float* __restrict__ raw, float* __restrict__ slots) {
    int tid = threadIdx.x;
    raw[tid] = 0.f;
    if (tid == 0) {
        ((int*)slots)[0] = f32_ord(-FLT_MAX);
        slots[1] = 0.f;
    }
}

// ---------------------------------------------------------------------------
// sc[n] = dot(tanh(feat[n,:] + qw), v) + neg(n); global max via ordered-int.
// ---------------------------------------------------------------------------
__global__ void __launch_bounds__(256)
scores_kernel(const float* __restrict__ feat, const float* __restrict__ qw,
              const float* __restrict__ v, const int* __restrict__ mask,
              float* __restrict__ sc, int* __restrict__ maxslot, int H) {
    __shared__ float qs[512];
    __shared__ float vs[512];
    __shared__ float wmax[8];
    int tid = threadIdx.x;
    for (int k = tid; k < H; k += 256) { qs[k] = qw[k]; vs[k] = v[k]; }
    __syncthreads();

    int wave = tid >> 5, lane = tid & 31;
    int n = blockIdx.x * 8 + wave;
    const float* row = feat + (size_t)n * H;
    float acc = 0.f;
    for (int i = 0; i < H / 32; ++i) {
        int k = lane + 32 * i;
        acc += tanhf(row[k] + qs[k]) * vs[k];
    }
    #pragma unroll
    for (int off = 16; off > 0; off >>= 1) acc += __shfl_xor(acc, off, 32);
    if (lane == 0) {
        float scn = acc + (mask[n] ? 0.f : NEG_INF_VAL);
        sc[n] = scn;
        wmax[wave] = scn;
    }
    __syncthreads();
    if (tid == 0) {
        float m = wmax[0];
        #pragma unroll
        for (int i = 1; i < 8; ++i) m = fmaxf(m, wmax[i]);
        atomicMax(maxslot, f32_ord(m));
    }
}

// ---------------------------------------------------------------------------
// w[n] = exp(sc[n] - max); accumulate denominator (one atomic per block).
// ---------------------------------------------------------------------------
__global__ void __launch_bounds__(256)
sumexp_kernel(const float* __restrict__ sc, float* __restrict__ w,
              float* __restrict__ slots) {
    __shared__ float red[256];
    int tid = threadIdx.x;
    int n = blockIdx.x * 256 + tid;
    float mx = ord_f32(((const int*)slots)[0]);
    float val = expf(sc[n] - mx);
    w[n] = val;
    red[tid] = val;
    __syncthreads();
    for (int s = 128; s > 0; s >>= 1) {
        if (tid < s) red[tid] += red[tid + s];
        __syncthreads();
    }
    if (tid == 0) atomicAdd(&slots[1], red[0]);
}

// ---------------------------------------------------------------------------
// raw[h] += sum_{n in chunk} w[n] * mem[n,h]; 64 rows/block, 2 atomics/thread.
// ---------------------------------------------------------------------------
__global__ void __launch_bounds__(256)
wsum_kernel(const float* __restrict__ w, const float* __restrict__ mem,
            float* __restrict__ raw, int H) {
    int tid = threadIdx.x;
    int n0 = blockIdx.x * 64;
    float acc0 = 0.f, acc1 = 0.f;
    for (int r = 0; r < 64; ++r) {
        float wn = w[n0 + r];
        const float* row = mem + (size_t)(n0 + r) * H;
        acc0 += wn * row[tid];
        acc1 += wn * row[tid + 256];
    }
    atomicAdd(&raw[tid], acc0);
    atomicAdd(&raw[tid + 256], acc1);
}

__global__ void normalize_kernel(const float* __restrict__ raw,
                                 const float* __restrict__ slots,
                                 float* __restrict__ dst) {
    int tid = threadIdx.x;
    dst[tid] = raw[tid] / slots[1];
}

// ---------------------------------------------------------------------------
// out[t] = dot(x, score_w) + score_b. One wave.
// ---------------------------------------------------------------------------
__global__ void score_kernel(const float* __restrict__ x, const float* __restrict__ sw,
                             const float* __restrict__ sb, float* __restrict__ out,
                             int D) {
    int lane = threadIdx.x;
    float acc = 0.f;
    for (int i = 0; i < D / 32; ++i) {
        int k = lane + 32 * i;
        acc += x[k] * sw[k];
    }
    #pragma unroll
    for (int off = 16; off > 0; off >>= 1) acc += __shfl_xor(acc, off, 32);
    if (lane == 0) out[0] = acc + sb[0];
}

extern "C" void kernel_launch(void* const* d_in, const int* in_sizes, int n_in,
                              void* d_out, int out_size, void* d_ws, size_t ws_size,
                              hipStream_t stream) {
    const float* attn_mem = (const float*)d_in[0];
    const float* attn_wm  = (const float*)d_in[1];
    const float* attn_wq  = (const float*)d_in[2];
    const float* attn_v   = (const float*)d_in[3];
    const float* hop_wm   = (const float*)d_in[4];
    const float* hop_wq   = (const float*)d_in[5];
    const float* hop_v    = (const float*)d_in[6];
    const float* init_i   = (const float*)d_in[7];
    const float* init_h   = (const float*)d_in[8];
    const float* init_c   = (const float*)d_in[9];
    const float* w_ih     = (const float*)d_in[10];
    const float* w_hh     = (const float*)d_in[11];
    const float* b_ih     = (const float*)d_in[12];
    const float* b_hh     = (const float*)d_in[13];
    const float* score_w  = (const float*)d_in[14];
    const float* score_b  = (const float*)d_in[15];
    const int*   mmask    = (const int*)d_in[16];

    const int N = in_sizes[16];      // 65536
    const int D = in_sizes[7];       // 512
    const int H = in_sizes[8];       // 512
    const int nstep = out_size;      // scores: (NSTEP,1,1) -> NSTEP
    float* out = (float*)d_out;

    // workspace layout (f32 elements)
    float* ws = (float*)d_ws;
    size_t fNH = (size_t)N * H;
    float* attn_feat = ws;                 // N*H
    float* hop_feat  = attn_feat + fNH;    // N*H
    float* sc    = hop_feat + fNH;         // N
    float* wgt   = sc + N;                 // N
    float* qw    = wgt + N;                // H
    float* raw   = qw + H;                 // H
    float* q2    = raw + H;                // H
    float* xbuf  = q2 + H;                 // D
    float* hb    = xbuf + D;               // 2*H (ping-pong)
    float* cb    = hb + 2 * H;             // 2*H (ping-pong)
    float* slots = cb + 2 * H;             // [0]=ordered-int max, [1]=sumexp

    // Fused precompute of attn_feat and hop_feat: one pass over attn_mem.
    dim3 ggrid(H / 128, N / 128);
    feat_gemm2_kernel<<<ggrid, 256, 0, stream>>>(attn_mem, attn_wm, hop_wm,
                                                 attn_feat, hop_feat, D, H);
    init_state_kernel<<<1, H, 0, stream>>>(init_i, init_h, init_c, xbuf, hb, cb);

    for (int t = 0; t < nstep; ++t) {
        const float* hin  = hb + (t & 1) * H;
        const float* cin  = cb + (t & 1) * H;
        float*       hout = hb + ((t + 1) & 1) * H;
        float*       cout = cb + ((t + 1) & 1) * H;

        lstm_kernel<<<1, H, 0, stream>>>(xbuf, hin, cin, w_ih, w_hh, b_ih, b_hh,
                                         hout, cout, D, H);

        // hop attention: query = h, feat = mem = hop_feat
        gemv_kernel<<<H / 256, 256, 0, stream>>>(hout, hop_wq, qw, H);
        attn_reset_kernel<<<1, H, 0, stream>>>(raw, slots);
        scores_kernel<<<N / 8, 256, 0, stream>>>(hop_feat, qw, hop_v, mmask, sc,
                                                 (int*)slots, H);
        sumexp_kernel<<<N / 256, 256, 0, stream>>>(sc, wgt, slots);
        wsum_kernel<<<N / 64, 256, 0, stream>>>(wgt, hop_feat, raw, H);
        normalize_kernel<<<1, H, 0, stream>>>(raw, slots, q2);

        // main attention: query = q2, feat = attn_feat, mem = attn_mem
        gemv_kernel<<<H / 256, 256, 0, stream>>>(q2, attn_wq, qw, H);
        attn_reset_kernel<<<1, H, 0, stream>>>(raw, slots);
        scores_kernel<<<N / 8, 256, 0, stream>>>(attn_feat, qw, attn_v, mmask, sc,
                                                 (int*)slots, H);
        sumexp_kernel<<<N / 256, 256, 0, stream>>>(sc, wgt, slots);
        wsum_kernel<<<N / 64, 256, 0, stream>>>(wgt, attn_mem, raw, H);
        normalize_kernel<<<1, H, 0, stream>>>(raw, slots, xbuf);

        score_kernel<<<1, 32, 0, stream>>>(xbuf, score_w, score_b, out + t, D);
    }
}
